// VectorQuantizer_24550033064210
// MI455X (gfx1250) — matrix operand order repaired and checked by reading, not compiled, not run on hardware
//
#include <hip/hip_runtime.h>
#include <hip/hip_bf16.h>

// ---------------------------------------------------------------------------
// VQ-VAE vector quantizer for gfx1250 (MI455X).
//   z:          [64, 256, 32, 32] f32   (d_in[0])
//   embeddings: [1024, 256]       f32   (d_in[1])
// Outputs (concatenated in d_out, f32):
//   z_q (==z_q_st numerically) [64,256,32,32], idx [64,32,32] (as float),
//   loss (scalar), perplexity (scalar)
//
// Distance GEMM runs as bf16 WMMA (f32 accumulate); codebook chunks are
// double-buffered in LDS via GLOBAL_LOAD_ASYNC_TO_LDS_B128 (ASYNCcnt),
// overlapping DMA with the WMMA chain.
// ---------------------------------------------------------------------------

#define B_DIM   64
#define C_DIM   256
#define HW      1024          // 32*32
#define NTOK    65536         // B*H*W
#define NE      1024          // codebook size
#define IDX_OFF  16777216
#define LOSS_OFF 16842752
#define PERP_OFF 16842753

#define LDS_PITCH 528         // 512B row + 16B pad (bank spread), multiple of 16
#define MC_CODES  32          // codes staged per macro chunk (double buffered)
#define N_MC      (NE / MC_CODES)
#define BUF_BYTES (MC_CODES * LDS_PITCH)

typedef __attribute__((ext_vector_type(16))) __bf16 v16bf;
typedef __attribute__((ext_vector_type(8)))  float  v8f;

union PackA { uint4 q[2]; v16bf v; };          // 16 bf16 = 2x b128 from LDS
union PackB { unsigned int u[8]; v16bf v; };   // 16 bf16 packed from f32 pairs

__device__ __forceinline__ unsigned int pack_bf16(float a, float b) {
  unsigned int ua = __float_as_uint(a), ub = __float_as_uint(b);
  ua = (ua + 0x7FFFu + ((ua >> 16) & 1u)) >> 16;   // round-to-nearest-even
  ub = (ub + 0x7FFFu + ((ub >> 16) & 1u)) >> 16;
  return ua | (ub << 16);
}

// ---------------------------------------------------------------------------
// Kernel 1: embeddings f32 -> bf16 (packed), ||e||^2, zero histogram
// grid: 1024 blocks (one per code) x 128 threads
// ---------------------------------------------------------------------------
__global__ __launch_bounds__(128)
void vq_prep(const float* __restrict__ emb, unsigned int* __restrict__ ebf,
             float* __restrict__ enorm, unsigned int* __restrict__ counts) {
  const int code = blockIdx.x;
  const int t = threadIdx.x;                 // handles channels 2t, 2t+1
  const float a = emb[(size_t)code * C_DIM + 2 * t];
  const float b = emb[(size_t)code * C_DIM + 2 * t + 1];
  ebf[(size_t)code * 128 + t] = pack_bf16(a, b);

  __shared__ float red[128];
  red[t] = a * a + b * b;
  __syncthreads();
  for (int s = 64; s > 0; s >>= 1) {
    if (t < s) red[t] += red[t + s];
    __syncthreads();
  }
  if (t == 0) {
    enorm[code] = red[0];
    counts[code] = 0u;
  }
}

// ---------------------------------------------------------------------------
// Kernel 2: fused distance-GEMM (bf16 WMMA) + argmin + gather + loss partials
// grid: 512 blocks x 256 threads (8 waves; each wave owns 16 tokens)
// ---------------------------------------------------------------------------
__global__ __launch_bounds__(256)
void vq_main(const float* __restrict__ z, const float* __restrict__ embF32,
             const unsigned int* __restrict__ ebf, const float* __restrict__ enorm,
             unsigned int* __restrict__ counts, float* __restrict__ lossPartial,
             float* __restrict__ out) {
  __shared__ __attribute__((aligned(16))) unsigned char ldsEb[2][BUF_BYTES];
  __shared__ __attribute__((aligned(16))) float ldsEn[2][MC_CODES];

  const int tid  = threadIdx.x;
  const int wave = tid >> 5;
  const int ln   = tid & 31;
  const int tl   = ln & 15;          // token lane within tile
  const int h    = ln >> 4;          // half-wave (K/code-half selector)
  const int gw   = blockIdx.x * 8 + wave;
  const int rowBase = gw * 16;       // global token base for this wave
  const int b  = rowBase >> 10;      // batch (tile never crosses a batch)
  const int tb = (rowBase & 1023) + tl;
  const size_t zBase = (size_t)b * (C_DIM * HW) + (size_t)tb;

  // --- Load the 16-token x 256-channel z tile once, as bf16 B-matrix tiles.
  // B layout (32x16, bf16): half h, V0-3 = K[h*8..h*8+7], V4-7 = K[16+h*8..],
  // VGPR v packs two consecutive K.
  PackB zt[8];
#pragma unroll
  for (int kc = 0; kc < 8; ++kc) {
#pragma unroll
    for (int v = 0; v < 8; ++v) {
      const int c0 = kc * 32 + h * 8 + 2 * v + (v >= 4 ? 8 : 0);
      const float f0 = z[zBase + (size_t)c0 * HW];
      const float f1 = z[zBase + (size_t)(c0 + 1) * HW];
      zt[kc].u[v] = pack_bf16(f0, f1);
    }
  }

  const uint4* ebf4 = (const uint4*)ebf;   // one uint4 = 8 bf16; 32 per code row

  // Async DMA stage of one 32-code chunk (32 x 512B rows) into LDS buffer.
  auto stage = [&](int mcIdx) {
    const int parity = mcIdx & 1;
    const int cb2 = mcIdx * MC_CODES;
    const unsigned dstBase = (unsigned)(uintptr_t)(void*)&ldsEb[parity][0];
#pragma unroll
    for (int k = 0; k < 4; ++k) {
      const int i = tid + k * 256;          // 0..1023 b128 transfers
      const int r = i >> 5, v4 = i & 31;
      const unsigned ldst = dstBase + r * LDS_PITCH + v4 * 16;
      const unsigned long long ga =
          (unsigned long long)(uintptr_t)(ebf4 + (size_t)(cb2 + r) * 32 + v4);
      asm volatile("global_load_async_to_lds_b128 %0, %1, off"
                   :: "v"(ldst), "v"(ga) : "memory");
    }
    if (tid < MC_CODES) ldsEn[parity][tid] = enorm[cb2 + tid];
  };

  float bestScore = 3.402823466e+38f;
  int   bestCode  = 0;

  stage(0);
  asm volatile("s_wait_asynccnt 0x0" ::: "memory");
  __syncthreads();

  for (int mc = 0; mc < N_MC; ++mc) {
    // Kick off the DMA for the next chunk; it lands while we do WMMAs.
    if (mc + 1 < N_MC) stage(mc + 1);

    const unsigned char* eb = &ldsEb[mc & 1][0];
    const float* en = &ldsEn[mc & 1][0];
    const int cb = mc * MC_CODES;

#pragma unroll
    for (int sub = 0; sub < 2; ++sub) {
      v8f acc = {0.f, 0.f, 0.f, 0.f, 0.f, 0.f, 0.f, 0.f};
      const int r0 = sub * 16 + tl;        // A-matrix row (code within chunk)
#pragma unroll
      for (int kc = 0; kc < 8; ++kc) {
        // A layout (16x32, bf16): half0 V0-3 = K[h*8..h*8+7], V4-7 = K[16+h*8..]
        PackA a;
        const unsigned char* p = eb + r0 * LDS_PITCH + kc * 64 + h * 16;
        a.q[0] = *(const uint4*)p;
        a.q[1] = *(const uint4*)(p + 32);
        acc = __builtin_amdgcn_wmma_f32_16x16x32_bf16(
            false, a.v, false, zt[kc].v, (short)0, acc, false, false);
      }
      // D layout: lane half h, VGPR j -> code cb + sub*16 + h*8 + j, token tl.
      const float4 en0 = *(const float4*)(en + sub * 16 + h * 8);
      const float4 en1 = *(const float4*)(en + sub * 16 + h * 8 + 4);
      const float ens[8] = {en0.x, en0.y, en0.z, en0.w, en1.x, en1.y, en1.z, en1.w};
      const int cbase = cb + sub * 16 + h * 8;
#pragma unroll
      for (int j = 0; j < 8; ++j) {
        const float sc = ens[j] - 2.0f * acc[j];   // ||e||^2 - 2 z.e
        if (sc < bestScore) { bestScore = sc; bestCode = cbase + j; }
      }
    }

    // Ensure this iteration's DMA landed and everyone is done reading the
    // buffer we will overwrite next iteration.
    asm volatile("s_wait_asynccnt 0x0" ::: "memory");
    __syncthreads();
  }

  // Merge the two code-halves held by lane pairs (ln, ln^16).
  const float os = __shfl_xor(bestScore, 16, 32);
  const int   oc = __shfl_xor(bestCode, 16, 32);
  if (os < bestScore || (os == bestScore && oc < bestCode)) {
    bestScore = os; bestCode = oc;
  }

  if (h == 0) {
    out[(size_t)IDX_OFF + rowBase + tl] = (float)bestCode;
    atomicAdd(&counts[bestCode], 1u);
  }

  // Gather z_q (== z_q_st numerically) and accumulate (z_q - z)^2.
  const float* er = embF32 + (size_t)bestCode * C_DIM;
  float lsum = 0.f;
#pragma unroll 4
  for (int i = 0; i < 128; ++i) {
    const int c = h * 128 + i;
    const float e  = er[c];
    const size_t off = zBase + (size_t)c * HW;
    const float zv = z[off];
    const float d  = e - zv;
    lsum += d * d;
    out[off] = e;
  }
  lsum += __shfl_xor(lsum, 1, 32);
  lsum += __shfl_xor(lsum, 2, 32);
  lsum += __shfl_xor(lsum, 4, 32);
  lsum += __shfl_xor(lsum, 8, 32);
  lsum += __shfl_xor(lsum, 16, 32);
  if (ln == 0) lossPartial[gw] = lsum;     // deterministic partial, no FP atomics
}

// ---------------------------------------------------------------------------
// Kernel 3: deterministic reductions -> loss, perplexity
// grid: 1 block x 256 threads
// ---------------------------------------------------------------------------
__global__ __launch_bounds__(256)
void vq_final(const unsigned int* __restrict__ counts,
              const float* __restrict__ lossPartial, float* __restrict__ out) {
  __shared__ float redE[256];
  __shared__ float redL[256];
  const int t = threadIdx.x;

  float ent = 0.f;
#pragma unroll
  for (int i = 0; i < 4; ++i) {
    const float p = (float)counts[t + i * 256] * (1.0f / 65536.0f);
    ent += p * logf(p + 1e-10f);
  }
  float ls = 0.f;
#pragma unroll
  for (int i = 0; i < 16; ++i) ls += lossPartial[t * 16 + i];

  redE[t] = ent; redL[t] = ls;
  __syncthreads();
  for (int s = 128; s > 0; s >>= 1) {
    if (t < s) { redE[t] += redE[t + s]; redL[t] += redL[t + s]; }
    __syncthreads();
  }
  if (t == 0) {
    // loss = beta*mean((sg(zq)-z)^2) + mean((zq-sg(z))^2) == 1.25*mean(diff^2)
    out[LOSS_OFF] = 1.25f * redL[0] * (1.0f / 16777216.0f);
    out[PERP_OFF] = expf(-redE[0]);
  }
}

// ---------------------------------------------------------------------------
extern "C" void kernel_launch(void* const* d_in, const int* in_sizes, int n_in,
                              void* d_out, int out_size, void* d_ws, size_t ws_size,
                              hipStream_t stream) {
  const float* z   = (const float*)d_in[0];
  const float* emb = (const float*)d_in[1];

  unsigned char* ws = (unsigned char*)d_ws;
  unsigned int* ebf        = (unsigned int*)ws;            // 1024*128 u32 (bf16 pairs) = 512 KB
  float*        enorm      = (float*)(ws + 524288);        // 1024 f32
  unsigned int* counts     = (unsigned int*)(ws + 528384); // 1024 u32
  float*        lossPart   = (float*)(ws + 532480);        // 4096 f32 (one per wave)
  float*        out        = (float*)d_out;

  vq_prep <<<NE, 128, 0, stream>>>(emb, ebf, enorm, counts);
  vq_main <<<NTOK / 128, 256, 0, stream>>>(z, emb, ebf, enorm, counts, lossPart, out);
  vq_final<<<1, 256, 0, stream>>>(counts, lossPart, out);
}
